// RNAEditingGNN_549755814553
// MI455X (gfx1250) — matrix operand order repaired and checked by reading, not compiled, not run on hardware
//
#include <hip/hip_runtime.h>
#include <hip/hip_bf16.h>

// ---------------------------------------------------------------------------
// GAT (3x GATConv + BN/ReLU) + mean-pool + FC + sigmoid, for MI455X (gfx1250).
// GEMMs use v_wmma_f32_16x16x32_f16 (wave32). Edge softmax via ordered-uint
// atomicMax + f32 atomicAdd. Aggregation is one-thread-per-(edge,channel) so
// the dominant gather/scatter pass is fully coalesced. All launches on
// `stream`; scratch carved from d_ws (graph-capture safe: no mallocs/syncs).
// ---------------------------------------------------------------------------

typedef __attribute__((ext_vector_type(16))) _Float16 v16h;
typedef __attribute__((ext_vector_type(8)))  _Float16 v8h;
typedef __attribute__((ext_vector_type(8)))  float    v8f;

#define HEADS 4
#define HID   16
#define CH    64            // HEADS*HID
#define NGRAPH 128
#define NEG_SLOPE 0.2f
#define BN_EPS 1e-5f

// ---- float <-> order-preserving u32 (for float atomic max via uint) --------
__device__ __forceinline__ unsigned f2ord(float f) {
  int i = __float_as_int(f);
  return (i >= 0) ? ((unsigned)i | 0x80000000u) : ~(unsigned)i;
}
__device__ __forceinline__ float ord2f(unsigned u) {
  int i = (u & 0x80000000u) ? (int)(u ^ 0x80000000u) : (int)~u;
  return __int_as_float(i);
}
#define ORD_NEG_INF 0x007FFFFFu   // f2ord(-inf)

// ---------------------------------------------------------------------------
__global__ void fill_u32_kernel(unsigned* __restrict__ p, unsigned v, long n) {
  long t = (long)blockIdx.x * blockDim.x + threadIdx.x;
  if (t < n) p[t] = v;
}

__global__ void cvt_f32_to_f16_kernel(const float* __restrict__ in,
                                      _Float16* __restrict__ out, long n) {
  long t = (long)blockIdx.x * blockDim.x + threadIdx.x;
  if (t < n) out[t] = (_Float16)in[t];
}

// ---------------------------------------------------------------------------
// GEMM: out[N,64] = A[N,K](f16) * W[64,K](f16)^T, f32 accumulate via WMMA.
// One wave -> 16 rows x 64 cols (4 head tiles). 4 waves / block.
// Fragment layouts per CDNA5 ISA 7.12.2 (wave32).
// ---------------------------------------------------------------------------
__global__ void gat_gemm_wmma_kernel(const _Float16* __restrict__ A,
                                     const _Float16* __restrict__ W,
                                     float* __restrict__ out,
                                     int nrows, int K) {
  const int wave = (blockIdx.x * blockDim.x + threadIdx.x) >> 5;
  const int lane = threadIdx.x & 31;
  const int ntile = (nrows + 15) >> 4;
  if (wave >= ntile) return;                 // wave-uniform exit (EXEC all-1 for WMMA)

  const int row0 = wave << 4;
  const int m  = lane & 15;
  const int hi = lane >> 4;
  // clamp A-row so partial tiles stay in-bounds (all lanes remain active)
  int arow = row0 + m; if (arow >= nrows) arow = nrows - 1;

  v8f c[4]; c[0] = v8f{}; c[1] = v8f{}; c[2] = v8f{}; c[3] = v8f{};

  for (int kk = 0; kk < K; kk += 32) {
    union { v16h v; v8h h[2]; } a;
    const _Float16* ap = A + (size_t)arow * K + kk + 8 * hi;
    a.h[0] = *(const v8h*)(ap);        // K = kk + 8*hi   .. +7   (elems 0..7)
    a.h[1] = *(const v8h*)(ap + 16);   // K = kk+16+8*hi  .. +7   (elems 8..15)
#pragma unroll
    for (int nt = 0; nt < 4; ++nt) {
      union { v16h v; v8h h[2]; } b;
      const _Float16* bp = W + (size_t)(nt * 16 + m) * K + kk + 8 * hi;
      b.h[0] = *(const v8h*)(bp);      // K = kk + 8*hi .. +7
      b.h[1] = *(const v8h*)(bp + 16); // K = kk+16+8*hi .. +7
      c[nt] = __builtin_amdgcn_wmma_f32_16x16x32_f16(
          /*neg_a=*/false, a.v, /*neg_b=*/false, b.v,
          /*c_mod=*/(short)0, c[nt], /*reuse_a=*/false, /*reuse_b=*/false);
    }
  }

#pragma unroll
  for (int r = 0; r < 8; ++r) {
    int row = row0 + r + 8 * hi;
    if (row < nrows) {
      float* o = out + (size_t)row * CH + m;
      o[0]  = c[0][r];
      o[16] = c[1][r];
      o[32] = c[2][r];
      o[48] = c[3][r];
    }
  }
}

// ---------------------------------------------------------------------------
// Per node-head attention coefficients: a_s = <h, att_src>, a_d = <h, att_dst>
// ---------------------------------------------------------------------------
__global__ void att_coef_kernel(const float* __restrict__ h,
                                const float* __restrict__ att_src,
                                const float* __restrict__ att_dst,
                                float* __restrict__ a_s, float* __restrict__ a_d,
                                int n) {
  int t = blockIdx.x * blockDim.x + threadIdx.x;
  if (t >= n * HEADS) return;
  int node = t >> 2, hd = t & 3;
  const float* hp = h + (size_t)node * CH + hd * HID;
  const float* s  = att_src + hd * HID;
  const float* dd = att_dst + hd * HID;
  float vs = 0.f, vd = 0.f;
#pragma unroll
  for (int c = 0; c < HID; ++c) { float v = hp[c]; vs += v * s[c]; vd += v * dd[c]; }
  a_s[t] = vs; a_d[t] = vd;
}

// ---------------------------------------------------------------------------
__device__ __forceinline__ void edge_sd(const int* __restrict__ ei, int E,
                                        int e, int& s, int& d) {
  if (e < E) { s = ei[e]; d = ei[E + e]; }
  else       { s = d = e - E; }             // self loops appended
}

__device__ __forceinline__ float edge_logit(const float* __restrict__ a_s,
                                            const float* __restrict__ a_d,
                                            int s, int d, int hd) {
  float l = a_s[s * HEADS + hd] + a_d[d * HEADS + hd];
  return (l > 0.f) ? l : NEG_SLOPE * l;     // leaky relu
}

__global__ void edge_max_kernel(const int* __restrict__ ei, int E, int Etot,
                                const float* __restrict__ a_s,
                                const float* __restrict__ a_d,
                                unsigned* __restrict__ mbuf) {
  int t = blockIdx.x * blockDim.x + threadIdx.x;
  if (t >= Etot * HEADS) return;
  int e = t >> 2, hd = t & 3, s, d;
  edge_sd(ei, E, e, s, d);
  float l = edge_logit(a_s, a_d, s, d, hd);
  atomicMax(&mbuf[d * HEADS + hd], f2ord(l));
}

// stores unnormalized exp per edge-head, accumulates z[dst,head]
__global__ void edge_sum_kernel(const int* __restrict__ ei, int E, int Etot,
                                const float* __restrict__ a_s,
                                const float* __restrict__ a_d,
                                const unsigned* __restrict__ mbuf,
                                float* __restrict__ zbuf,
                                float* __restrict__ exbuf) {
  int t = blockIdx.x * blockDim.x + threadIdx.x;
  if (t >= Etot * HEADS) return;
  int e = t >> 2, hd = t & 3, s, d;
  edge_sd(ei, E, e, s, d);
  float l  = edge_logit(a_s, a_d, s, d, hd);
  float ex = __expf(l - ord2f(mbuf[d * HEADS + hd]));
  exbuf[t] = ex;
  atomicAdd(&zbuf[d * HEADS + hd], ex);
}

// One thread per (edge, channel): fully-coalesced row gather + scatter.
// acc[d, c] += hlin[s, c] * alpha(e, c/16);  lane c%16==0 writes alpha0.
__global__ void edge_aggregate_kernel(const int* __restrict__ ei, int E, long nwork,
                                      const float* __restrict__ exbuf,
                                      const float* __restrict__ zbuf,
                                      const float* __restrict__ hlin,
                                      float* __restrict__ acc,
                                      float* __restrict__ alpha_out) {
  long t = (long)blockIdx.x * blockDim.x + threadIdx.x;
  if (t >= nwork) return;
  int e  = (int)(t >> 6);
  int c  = (int)(t & (CH - 1));
  int hd = c >> 4;
  int s, d;
  edge_sd(ei, E, e, s, d);
  float alpha = exbuf[e * HEADS + hd] / zbuf[d * HEADS + hd];
  if (alpha_out && (c & 15) == 0) alpha_out[(size_t)e * HEADS + hd] = alpha;
  atomicAdd(&acc[(size_t)d * CH + c], hlin[(size_t)s * CH + c] * alpha);
}

// ---------------------------------------------------------------------------
// bias + ReLU + BatchNorm (eval); write f32 in-place and f16 copy for next GEMM
// ---------------------------------------------------------------------------
__global__ void bias_relu_bn_kernel(float* __restrict__ acc,
                                    _Float16* __restrict__ h16,
                                    const float* __restrict__ bias,
                                    const float* __restrict__ gamma,
                                    const float* __restrict__ beta,
                                    const float* __restrict__ mean,
                                    const float* __restrict__ var,
                                    int n) {
  int t = blockIdx.x * blockDim.x + threadIdx.x;
  if (t >= n * CH) return;
  int c = t & (CH - 1);
  float v = acc[t] + bias[c];
  v = v > 0.f ? v : 0.f;
  v = gamma[c] * (v - mean[c]) * rsqrtf(var[c] + BN_EPS) + beta[c];
  acc[t] = v;
  h16[t] = (_Float16)v;
}

// ---------------------------------------------------------------------------
__global__ void pool_accum_kernel(const float* __restrict__ h,
                                  const int* __restrict__ batch,
                                  float* __restrict__ psum,
                                  float* __restrict__ pcnt, int n) {
  int t = blockIdx.x * blockDim.x + threadIdx.x;
  if (t >= n * CH) return;
  int node = t >> 6, c = t & (CH - 1);
  int g = batch[node];
  atomicAdd(&psum[g * CH + c], h[t]);
  if (c == 0) atomicAdd(&pcnt[g], 1.0f);
}

__global__ void final_fc_kernel(const float* __restrict__ psum,
                                const float* __restrict__ pcnt,
                                const float* __restrict__ fcW,
                                const float* __restrict__ fcb,
                                float* __restrict__ out) {
  int g = blockIdx.x * blockDim.x + threadIdx.x;
  if (g >= NGRAPH) return;
  float cnt = pcnt[g]; cnt = cnt > 1.0f ? cnt : 1.0f;
  float acc = fcb[0];
#pragma unroll
  for (int c = 0; c < CH; ++c) acc += (psum[g * CH + c] / cnt) * fcW[c];
  out[g] = 1.0f / (1.0f + __expf(-acc));
}

// ---------------------------------------------------------------------------
extern "C" void kernel_launch(void* const* d_in, const int* in_sizes, int n_in,
                              void* d_out, int out_size, void* d_ws, size_t ws_size,
                              hipStream_t stream) {
  const int IN_DIM = 128;
  const int N = in_sizes[0] / IN_DIM;      // 50000
  const int E = in_sizes[1] / 2;           // 800000
  const int Etot = E + N;                  // self loops appended

  const float* x      = (const float*)d_in[0];
  const int*   ei     = (const int*)d_in[1];
  const int*   batch  = (const int*)d_in[2];
  const float* fcW    = (const float*)d_in[27];
  const float* fcb    = (const float*)d_in[28];

  // ---- carve workspace --------------------------------------------------
  char* p = (char*)d_ws;
  auto carve = [&](size_t bytes) -> void* {
    void* r = (void*)p; p += (bytes + 255) & ~(size_t)255; return r;
  };
  _Float16* xf16  = (_Float16*)carve((size_t)N * IN_DIM * 2);
  _Float16* hf16  = (_Float16*)carve((size_t)N * CH * 2);
  _Float16* wf16  = (_Float16*)carve((size_t)CH * IN_DIM * 2);
  float*    hlin  = (float*)carve((size_t)N * CH * 4);
  float*    hacc  = (float*)carve((size_t)N * CH * 4);   // agg accumulator / BN out
  float*    asrc  = (float*)carve((size_t)N * HEADS * 4);
  float*    adst  = (float*)carve((size_t)N * HEADS * 4);
  unsigned* mbuf  = (unsigned*)carve((size_t)N * HEADS * 4);
  float*    zbuf  = (float*)carve((size_t)N * HEADS * 4);
  float*    exbuf = (float*)carve((size_t)Etot * HEADS * 4);
  float*    psum  = (float*)carve((size_t)NGRAPH * CH * 4);
  float*    pcnt  = (float*)carve((size_t)NGRAPH * 4);

  const int TB = 256;
  auto blk = [](long n, int t) { return (int)((n + t - 1) / t); };

  // x -> f16 once
  cvt_f32_to_f16_kernel<<<blk((long)N * IN_DIM, TB), TB, 0, stream>>>(x, xf16, (long)N * IN_DIM);

  float* alpha0_out = (float*)d_out + NGRAPH;   // outputs: [out(128), alpha0(Etot*4)]

  for (int l = 0; l < 3; ++l) {
    const int d = (l == 0) ? IN_DIM : CH;
    const float* W       = (const float*)d_in[3 + 8 * l + 0];
    const float* att_src = (const float*)d_in[3 + 8 * l + 1];
    const float* att_dst = (const float*)d_in[3 + 8 * l + 2];
    const float* bias    = (const float*)d_in[3 + 8 * l + 3];
    const float* gamma   = (const float*)d_in[3 + 8 * l + 4];
    const float* beta    = (const float*)d_in[3 + 8 * l + 5];
    const float* mean    = (const float*)d_in[3 + 8 * l + 6];
    const float* var     = (const float*)d_in[3 + 8 * l + 7];

    // weights -> f16
    cvt_f32_to_f16_kernel<<<blk(CH * d, TB), TB, 0, stream>>>(W, wf16, CH * d);

    // GEMM (WMMA): hlin = A @ W^T
    const _Float16* Ain = (l == 0) ? xf16 : hf16;
    const int ntile = (N + 15) / 16;
    gat_gemm_wmma_kernel<<<blk(ntile, 4), 128, 0, stream>>>(Ain, wf16, hlin, N, d);

    // attention coefficients
    att_coef_kernel<<<blk((long)N * HEADS, TB), TB, 0, stream>>>(
        hlin, att_src, att_dst, asrc, adst, N);

    // init softmax buffers + aggregation accumulator
    fill_u32_kernel<<<blk((long)N * HEADS, TB), TB, 0, stream>>>(mbuf, ORD_NEG_INF, (long)N * HEADS);
    fill_u32_kernel<<<blk((long)N * HEADS, TB), TB, 0, stream>>>((unsigned*)zbuf, 0u, (long)N * HEADS);
    fill_u32_kernel<<<blk((long)N * CH, TB), TB, 0, stream>>>((unsigned*)hacc, 0u, (long)N * CH);

    // scatter softmax
    const long eh = (long)Etot * HEADS;
    edge_max_kernel<<<blk(eh, TB), TB, 0, stream>>>(ei, E, Etot, asrc, adst, mbuf);
    edge_sum_kernel<<<blk(eh, TB), TB, 0, stream>>>(ei, E, Etot, asrc, adst, mbuf, zbuf, exbuf);

    // coalesced weighted aggregation (one thread per edge-channel)
    const long ew = (long)Etot * CH;
    edge_aggregate_kernel<<<blk(ew, TB), TB, 0, stream>>>(
        ei, E, ew, exbuf, zbuf, hlin, hacc,
        (l == 0) ? alpha0_out : nullptr);

    // bias + relu + batchnorm -> hacc (f32) + hf16 (next layer A)
    bias_relu_bn_kernel<<<blk((long)N * CH, TB), TB, 0, stream>>>(
        hacc, hf16, bias, gamma, beta, mean, var, N);
  }

  // global mean pool + FC + sigmoid
  fill_u32_kernel<<<blk(NGRAPH * CH, TB), TB, 0, stream>>>((unsigned*)psum, 0u, NGRAPH * CH);
  fill_u32_kernel<<<1, TB, 0, stream>>>((unsigned*)pcnt, 0u, NGRAPH);
  pool_accum_kernel<<<blk((long)N * CH, TB), TB, 0, stream>>>(hacc, batch, psum, pcnt, N);
  final_fc_kernel<<<1, NGRAPH, 0, stream>>>(psum, pcnt, fcW, fcb, (float*)d_out);
}
